// IIF_78743930405236
// MI455X (gfx1250) — matrix operand order repaired and checked by reading, not compiled, not run on hardware
//
#include <hip/hip_runtime.h>
#include <hip/hip_bf16.h>

typedef __attribute__((ext_vector_type(8)))  _Float16 v8h;
typedef __attribute__((ext_vector_type(16))) _Float16 v16h;
typedef __attribute__((ext_vector_type(8)))  float    v8f;

#define B_SZ   2
#define N_PTS  30720
#define H_LR   128
#define H_HR   512
#define CCH    128        // feat/guide channels
#define NCLS   19

// padded layer dims
#define K0P 416           // 386 -> 416 (13 k-tiles of 32)
#define N0  512
#define N1  256
#define N2  128
#define N3  64
#define N4P 32            // 20 -> 32

// per-wave LDS: X slab 16x416 f16 (13312 B) + H slab 16x512 f16 (16384 B)
#define X_BYTES   (16 * K0P * 2)
#define H_BYTES   (16 * N0 * 2)
#define PER_WAVE  (X_BYTES + H_BYTES)      // 29696
#define WAVES_PER_BLK 4
#define SMEM_BYTES (PER_WAVE * WAVES_PER_BLK)  // 118784

// ---------------------------------------------------------------------------
// Transpose [B,C,H,W] f32 -> [B,H,W,C] f16, LDS-tiled (32x32 in (c,x) plane)
// ---------------------------------------------------------------------------
__global__ __launch_bounds__(256) void transpose_chw_hwc(
    const float* __restrict__ in, _Float16* __restrict__ out, int C, int H, int W)
{
    __shared__ float tile[32][33];
    const int bh = blockIdx.z;            // b*H + y
    const int b  = bh / H;
    const int y  = bh - b * H;
    const int x0 = blockIdx.x * 32;
    const int c0 = blockIdx.y * 32;
    const int tx = threadIdx.x;           // 0..31
    const int ty = threadIdx.y;           // 0..7
#pragma unroll
    for (int i = 0; i < 4; ++i) {
        int ch = c0 + ty + 8 * i;
        tile[ty + 8 * i][tx] = in[(((size_t)b * C + ch) * H + y) * W + x0 + tx];
    }
    __syncthreads();
#pragma unroll
    for (int i = 0; i < 4; ++i) {
        int x = x0 + ty + 8 * i;
        out[(((size_t)b * H + y) * W + x) * C + c0 + tx] = (_Float16)tile[tx][ty + 8 * i];
    }
}

// ---------------------------------------------------------------------------
// Pre-swizzle a weight matrix [Kin x Nin] (row-major f32) into WMMA B-fragment
// order, f16, zero-padded to [Kp x Np]. Per 32x16 (KxN) tile: 512 halves,
// index = tile*512 + lane*16 + h ; k = kt*32 + 16*(lane>>4) + h ; n = nt*16+(lane&15)
// (contiguous k order used here, same as load_afrag: lanes 0-15 hold K=0..15, lanes 16-31 K=16..31)
// ---------------------------------------------------------------------------
__global__ void prep_weights(const float* __restrict__ w, _Float16* __restrict__ out,
                             int Kin, int Nin, int Np, int total)
{
    int tid = blockIdx.x * 256 + threadIdx.x;
    if (tid >= total) return;
    int tile   = tid >> 9;
    int within = tid & 511;
    int lane   = within >> 4;
    int h      = within & 15;
    int Ntiles = Np >> 4;
    int kt = tile / Ntiles;
    int nt = tile - kt * Ntiles;
    int k  = kt * 32 + ((lane >> 4) << 4) + h;
    int n  = nt * 16 + (lane & 15);
    float v = (k < Kin && n < Nin) ? w[(size_t)k * Nin + n] : 0.0f;
    out[tid] = (_Float16)v;
}

// ---------------------------------------------------------------------------
// Fragment loaders
// A (16-bit 16x32): lane l -> row M=l&15; halves 0..15 = K base..base+15,
// base = kt*32 + 16*(l>>4) (matches B fragment order). Row-major LDS ->
// two contiguous 16B chunks (ds_load_b128 each).
// ---------------------------------------------------------------------------
__device__ __forceinline__ v16h load_afrag(const _Float16* A, int stride, int kt, int lane)
{
    const _Float16* ap = A + (lane & 15) * stride + kt * 32 + ((lane >> 4) << 4);
    union { v16h v; v8h h[2]; } a;
    a.h[0] = *(const v8h*)ap;
    a.h[1] = *(const v8h*)(ap + 8);
    return a.v;
}

__device__ __forceinline__ v16h load_bfrag(const _Float16* Wf, int tile, int lane)
{
    const _Float16* wp = Wf + (size_t)tile * 512 + lane * 16;
    union { v16h v; v8h h[2]; } b;
    b.h[0] = *(const v8h*)wp;
    b.h[1] = *(const v8h*)(wp + 8);
    return b.v;
}

// one hidden layer: O = relu(A @ W + bias), f16 out to LDS
__device__ __forceinline__ void mlp_layer(const _Float16* A, int As,
                                          const _Float16* __restrict__ Wf,
                                          const float* __restrict__ bias,
                                          _Float16* O, int Os,
                                          int Kt, int Nt, int lane)
{
    const int ncol  = lane & 15;
    const int rbase = (lane >> 4) * 8;
    for (int nt = 0; nt < Nt; ++nt) {
        float bv = bias[nt * 16 + ncol];
        v8f c = {bv, bv, bv, bv, bv, bv, bv, bv};
        for (int kt = 0; kt < Kt; ++kt) {
            v16h a = load_afrag(A, As, kt, lane);
            v16h b = load_bfrag(Wf, kt * Nt + nt, lane);
            c = __builtin_amdgcn_wmma_f32_16x16x32_f16(false, a, false, b,
                                                       (short)0, c, false, false);
        }
        const int n = nt * 16 + ncol;
#pragma unroll
        for (int i = 0; i < 8; ++i) {
            float x = c[i];
            x = x > 0.0f ? x : 0.0f;
            O[(rbase + i) * Os + n] = (_Float16)x;
        }
    }
}

// ---------------------------------------------------------------------------
// Fused gather + 5-layer MLP + softmax blend.
// grid: 3840 blocks x 128 threads (4 waves). Wave w owns row-block
// t = blockIdx*4+w -> rows 16t..16t+15 ; global row R = (point<<2)|corner.
// ---------------------------------------------------------------------------
__global__ __launch_bounds__(128) void fused_mlp(
    const float* __restrict__ coord,
    const _Float16* __restrict__ featT,   // [B,128,128,128] HWC f16
    const _Float16* __restrict__ hrT,     // [B,512,512,128]
    const _Float16* __restrict__ lrT,     // [B,128,128,128]
    const _Float16* __restrict__ W0f, const _Float16* __restrict__ W1f,
    const _Float16* __restrict__ W2f, const _Float16* __restrict__ W3f,
    const _Float16* __restrict__ W4f,
    const float* __restrict__ b0, const float* __restrict__ b1,
    const float* __restrict__ b2, const float* __restrict__ b3,
    const float* __restrict__ b4,
    float* __restrict__ out)
{
    extern __shared__ char smemRaw[];
    const int lane = threadIdx.x & 31;
    const int wave = threadIdx.x >> 5;
    _Float16* X  = (_Float16*)(smemRaw + wave * PER_WAVE);            // 16 x 416
    _Float16* Ha = (_Float16*)(smemRaw + wave * PER_WAVE + X_BYTES);  // 16 x 512
    const int t = blockIdx.x * WAVES_PER_BLK + wave;

    // ---------------- gather phase: build X[16][416] (f16) ----------------
    for (int r = 0; r < 16; ++r) {
        const int R = t * 16 + r;
        const int pI = R >> 2;
        const int corner = R & 3;
        const int b = pI / N_PTS;
        const float cy = coord[(size_t)pI * 2 + 0];
        const float cx = coord[(size_t)pI * 2 + 1];

        // hr sample at `coord` (shared by all 4 corners of the point)
        int iyH = (int)rintf(((cy + 1.0f) * 512.0f - 1.0f) * 0.5f);
        int ixH = (int)rintf(((cx + 1.0f) * 512.0f - 1.0f) * 0.5f);
        bool vH = (iyH >= 0) && (iyH < 512) && (ixH >= 0) && (ixH < 512);
        int iyHc = min(max(iyH, 0), 511), ixHc = min(max(ixH, 0), 511);

        // corner-shifted coordinate
        const float vx = (corner & 2) ? 1.0f : -1.0f;
        const float vy = (corner & 1) ? 1.0f : -1.0f;
        const float cyc = cy + vx * (1.0f / 128.0f);
        const float cxc = cx + vy * (1.0f / 128.0f);
        int iy = (int)rintf(((cyc + 1.0f) * 128.0f - 1.0f) * 0.5f);
        int ix = (int)rintf(((cxc + 1.0f) * 128.0f - 1.0f) * 0.5f);
        bool v = (iy >= 0) && (iy < 128) && (ix >= 0) && (ix < 128);
        int iyc = min(max(iy, 0), 127), ixc = min(max(ix, 0), 127);

        union U { uint2 u; _Float16 h[4]; };
        U qf, qh, ql, qd;
        qf.u = make_uint2(0u, 0u); qh.u = qf.u; ql.u = qf.u;
        const size_t lrBase = ((size_t)(b * 128 + iyc) * 128 + ixc) * CCH;
        const size_t hrBase = ((size_t)(b * 512 + iyHc) * 512 + ixHc) * CCH;
        if (v)  qf.u = *(const uint2*)(featT + lrBase + lane * 4);
        if (vH) qh.u = *(const uint2*)(hrT   + hrBase + lane * 4);
        if (v)  ql.u = *(const uint2*)(lrT   + lrBase + lane * 4);
#pragma unroll
        for (int i = 0; i < 4; ++i)
            qd.h[i] = (_Float16)((float)qh.h[i] - (float)ql.h[i]);

        _Float16* Xr = X + r * K0P;
        *(uint2*)(Xr +       lane * 4) = qf.u;   // q_feat        [0..127]
        *(uint2*)(Xr + 128 + lane * 4) = qh.u;   // q_guide_hr    [128..255]
        *(uint2*)(Xr + 256 + lane * 4) = qd.u;   // hr - lr       [256..383]
        if (lane == 0) {
            float qcy = v ? ((2 * iyc + 1) * (1.0f / 128.0f) - 1.0f) : 0.0f;
            float qcx = v ? ((2 * ixc + 1) * (1.0f / 128.0f) - 1.0f) : 0.0f;
            Xr[384] = (_Float16)((cy - qcy) * 128.0f);
            Xr[385] = (_Float16)((cx - qcx) * 128.0f);
        } else if (lane < 16) {                  // zero pad [386..415]
            Xr[384 + lane * 2] = (_Float16)0.0f;
            Xr[385 + lane * 2] = (_Float16)0.0f;
        }
    }

    // ---------------- MLP: ping-pong X <-> Ha (wave-local, no barriers) ----
    mlp_layer(X,  K0P, W0f, b0, Ha, N0, K0P / 32, N0 / 16, lane);  // 386p->512
    mlp_layer(Ha, N0,  W1f, b1, X,  N1, N0 / 32,  N1 / 16, lane);  // 512->256
    mlp_layer(X,  N1,  W2f, b2, Ha, N2, N1 / 32,  N2 / 16, lane);  // 256->128
    mlp_layer(Ha, N2,  W3f, b3, X,  N3, N2 / 32,  N3 / 16, lane);  // 128->64

    // final layer: 64 -> 32 (20 real), f32 result into Ha as float[16][32]
    float* E = (float*)Ha;
    {
        const int ncol  = lane & 15;
        const int rbase = (lane >> 4) * 8;
        for (int nt = 0; nt < 2; ++nt) {
            const int n = nt * 16 + ncol;
            float bv = (n < 20) ? b4[n] : 0.0f;
            v8f c = {bv, bv, bv, bv, bv, bv, bv, bv};
            for (int kt = 0; kt < 2; ++kt) {
                v16h a = load_afrag(X, N3, kt, lane);
                v16h b = load_bfrag(W4f, kt * 2 + nt, lane);
                c = __builtin_amdgcn_wmma_f32_16x16x32_f16(false, a, false, b,
                                                           (short)0, c, false, false);
            }
#pragma unroll
            for (int i = 0; i < 8; ++i)
                E[(rbase + i) * N4P + n] = c[i];
        }
    }

    // ---------------- epilogue: softmax over 4 corners, blend, store -------
    for (int q = 0; q < 4; ++q) {
        float l0 = E[(q * 4 + 0) * N4P + NCLS];
        float l1 = E[(q * 4 + 1) * N4P + NCLS];
        float l2 = E[(q * 4 + 2) * N4P + NCLS];
        float l3 = E[(q * 4 + 3) * N4P + NCLS];
        float m  = fmaxf(fmaxf(l0, l1), fmaxf(l2, l3));
        float e0 = __expf(l0 - m), e1 = __expf(l1 - m);
        float e2 = __expf(l2 - m), e3 = __expf(l3 - m);
        float inv = 1.0f / (e0 + e1 + e2 + e3);
        if (lane < NCLS) {
            float o = e0 * E[(q * 4 + 0) * N4P + lane]
                    + e1 * E[(q * 4 + 1) * N4P + lane]
                    + e2 * E[(q * 4 + 2) * N4P + lane]
                    + e3 * E[(q * 4 + 3) * N4P + lane];
            const int p = t * 4 + q;          // global point = b*N_PTS+n
            out[(size_t)p * NCLS + lane] = o * inv;
        }
    }
}

// ---------------------------------------------------------------------------
extern "C" void kernel_launch(void* const* d_in, const int* in_sizes, int n_in,
                              void* d_out, int out_size, void* d_ws, size_t ws_size,
                              hipStream_t stream)
{
    const float* feat  = (const float*)d_in[0];
    const float* coord = (const float*)d_in[1];
    const float* hr    = (const float*)d_in[2];
    const float* lr    = (const float*)d_in[3];
    const float* w[5]  = {(const float*)d_in[4],  (const float*)d_in[6],
                          (const float*)d_in[8],  (const float*)d_in[10],
                          (const float*)d_in[12]};
    const float* bias[5] = {(const float*)d_in[5],  (const float*)d_in[7],
                            (const float*)d_in[9],  (const float*)d_in[11],
                            (const float*)d_in[13]};

    // workspace layout (f16), 256B aligned
    char* ws = (char*)d_ws;
    size_t off = 0;
    auto alloc = [&](size_t bytes) { size_t o = off; off = (off + bytes + 255) & ~(size_t)255; return o; };
    _Float16* featT = (_Float16*)(ws + alloc((size_t)B_SZ * 128 * 128 * CCH * 2));
    _Float16* lrT   = (_Float16*)(ws + alloc((size_t)B_SZ * 128 * 128 * CCH * 2));
    _Float16* hrT   = (_Float16*)(ws + alloc((size_t)B_SZ * 512 * 512 * CCH * 2));
    const int Kp[5] = {K0P, 512, 256, 128, 64};
    const int Np[5] = {N0,  N1,  N2,  N3,  N4P};
    const int Kin[5] = {386, 512, 256, 128, 64};
    const int Nin[5] = {512, 256, 128, 64,  20};
    _Float16* Wf[5];
    for (int i = 0; i < 5; ++i)
        Wf[i] = (_Float16*)(ws + alloc((size_t)Kp[i] * Np[i] * 2));

    // transposes
    transpose_chw_hwc<<<dim3(128 / 32, CCH / 32, B_SZ * 128), dim3(32, 8), 0, stream>>>(
        feat, featT, CCH, 128, 128);
    transpose_chw_hwc<<<dim3(128 / 32, CCH / 32, B_SZ * 128), dim3(32, 8), 0, stream>>>(
        lr, lrT, CCH, 128, 128);
    transpose_chw_hwc<<<dim3(512 / 32, CCH / 32, B_SZ * 512), dim3(32, 8), 0, stream>>>(
        hr, hrT, CCH, 512, 512);

    // weight swizzle
    for (int i = 0; i < 5; ++i) {
        int total = Kp[i] * Np[i];
        prep_weights<<<(total + 255) / 256, 256, 0, stream>>>(w[i], Wf[i], Kin[i], Nin[i], Np[i], total);
    }

    // fused MLP: 245760 rows / (16 rows * 4 waves) = 3840 blocks
    (void)hipFuncSetAttribute((const void*)fused_mlp,
                              hipFuncAttributeMaxDynamicSharedMemorySize, SMEM_BYTES);
    fused_mlp<<<3840, 128, SMEM_BYTES, stream>>>(
        coord, featT, hrT, lrT,
        Wf[0], Wf[1], Wf[2], Wf[3], Wf[4],
        bias[0], bias[1], bias[2], bias[3], bias[4],
        (float*)d_out);
    (void)in_sizes; (void)n_in; (void)out_size; (void)ws_size;
}
